// Model_11879879543234
// MI455X (gfx1250) — matrix operand order repaired and checked by reading, not compiled, not run on hardware
//
#include <hip/hip_runtime.h>

// ---------------- types ----------------
typedef __attribute__((ext_vector_type(16))) __bf16  v16bf;
typedef __attribute__((ext_vector_type(8)))  float   v8f;
typedef __attribute__((ext_vector_type(4)))  unsigned int v4u;
typedef __attribute__((ext_vector_type(4)))  int v4i;

#define B_ 32
#define S_ 512
#define E_ 512
#define H_ 1024
#define G_ 3072   // 3*H

__device__ __forceinline__ unsigned short f2bf(float f) {
  unsigned int u = __builtin_bit_cast(unsigned int, f);
  u += 0x7FFFu + ((u >> 16) & 1u);          // round-to-nearest-even
  return (unsigned short)(u >> 16);
}

union ABv { v4u u[2]; v16bf v; };

// ---------------- init: zero recurrent state + sync counter ----------------
__global__ void init_kernel(float* h_f32, unsigned short* h_bf, int* sync) {
  int i = blockIdx.x * 256 + threadIdx.x;   // 128 blocks * 256 = 32768 = B*H
  h_f32[i] = 0.f;
  h_bf[i]  = 0;
  if (i == 0) *sync = 0;
}

// ---------------- swizzle f32 [K,N] -> bf16 WMMA-B-fragment blocked layout ----
// Tile (kt,nt) of 32(K)x16(N) stored as 512 contiguous bf16:
//   lane l (0..31) owns 16 values at tile_base + l*16:
//   value kk (0..15): K = kt*32 + (l>=16?16:0) + kk, N = nt*16 + (l&15)
// Tiles ordered (nt * (K/32) + kt).
__global__ void swizzle_kernel(const float* __restrict__ src,
                               unsigned short* __restrict__ dst, int K, int N) {
  int idx = blockIdx.x * 256 + threadIdx.x;
  int Kt = K >> 5;
  int tile   = idx >> 9;
  int within = idx & 511;
  int lane = within >> 4;
  int kk   = within & 15;
  int nt = tile / Kt;
  int kt = tile - nt * Kt;
  int k = kt * 32 + ((lane >> 4) << 4) + kk;
  int n = nt * 16 + (lane & 15);
  dst[idx] = f2bf(src[(size_t)k * N + n]);
}

// ---------------- xw = emb[x] @ W + b_i  (bf16 WMMA, f32 accumulate) --------
// grid (1024, 24): blockIdx.x = M-tile (16 rows of B*S), blockIdx.y*8+wave = N-tile
__global__ void __launch_bounds__(256) xw_kernel(const int* __restrict__ x,
                                                 const float* __restrict__ emb,
                                                 const unsigned short* __restrict__ Wsw,
                                                 const float* __restrict__ b_i,
                                                 float* __restrict__ xw) {
  __shared__ __align__(16) unsigned short sA[16 * E_];   // 16 rows x 512, 16 KB
  const int tid = threadIdx.x;
  const int mt  = blockIdx.x;
  // stage 16 gathered embedding rows -> LDS as bf16
  {
    int r   = tid >> 4;
    int seg = tid & 15;
    int tok = x[mt * 16 + r];
    const float4* erow = (const float4*)(emb + (size_t)tok * E_) + seg * 8;
    unsigned short* d = &sA[r * E_ + seg * 32];
    #pragma unroll
    for (int j = 0; j < 8; ++j) {
      float4 f = erow[j];
      d[j*4+0] = f2bf(f.x); d[j*4+1] = f2bf(f.y);
      d[j*4+2] = f2bf(f.z); d[j*4+3] = f2bf(f.w);
    }
  }
  __syncthreads();

  const int wave = tid >> 5, lane = tid & 31;
  const int nt = blockIdx.y * 8 + wave;
  const int laneHi = lane >> 4, lane15 = lane & 15;
  const v4u* wb = (const v4u*)(Wsw + (size_t)nt * 16 * 512) + (lane << 1);

  v8f c = {};
  #pragma unroll 4
  for (int kt = 0; kt < E_ / 32; ++kt) {
    ABv a, b;
    const v4u* ap = (const v4u*)&sA[lane15 * E_ + kt * 32 + (laneHi << 4)];
    a.u[0] = ap[0];               // K base..base+7
    a.u[1] = ap[1];               // K base+8..base+15
    const v4u* bp = wb + kt * 64; // 512 bf16 per tile = 64 v4u
    b.u[0] = bp[0];
    b.u[1] = bp[1];
    c = __builtin_amdgcn_wmma_f32_16x16x32_bf16(false, a.v, false, b.v,
                                                (short)0, c, false, false);
  }
  const float bias = b_i[nt * 16 + lane15];
  const int grow = mt * 16 + (laneHi << 3);
  const int gcol = nt * 16 + lane15;
  #pragma unroll
  for (int r = 0; r < 8; ++r)
    xw[(size_t)(grow + r) * G_ + gcol] = c[r] + bias;
}

// ---------------- persistent GRU scan -------------------------------------
#if __has_builtin(__builtin_amdgcn_global_load_async_to_lds_b128) && \
    __has_builtin(__builtin_amdgcn_s_wait_asynccnt)
#define ASYNC_LDS_OK 1
__device__ __forceinline__ void async_copy_b128(const v4u* g, v4u* l) {
  __builtin_amdgcn_global_load_async_to_lds_b128(
      (__attribute__((address_space(1))) v4i*)g,
      (__attribute__((address_space(3))) v4i*)l, 0, 0);
}
#else
#define ASYNC_LDS_OK 0
#endif

__global__ void __launch_bounds__(256) gru_scan_kernel(
    const float* __restrict__ xw,
    const unsigned short* __restrict__ Usw,
    const float* __restrict__ b_h,
    float* __restrict__ h_f32,
    unsigned short* __restrict__ h_bf,
    int* syncCnt,
    float* __restrict__ out) {
  __shared__ __align__(16) unsigned short sH[B_ * H_];   // 64 KB bf16 state
  const int tid  = threadIdx.x;
  const int bk   = blockIdx.x;        // 0..15, owns H-cols [bk*64, bk*64+64)
  const int wave = tid >> 5, lane = tid & 31;
  const int m    = wave >> 2;         // 0..1  -> rows m*16..m*16+15
  const int nc   = wave & 3;          // 0..3  -> 16-col group
  const int laneHi = lane >> 4, lane15 = lane & 15;
  const int jcol = bk * 64 + nc * 16 + lane15;
  const float bhz = b_h[jcol];
  const float bhr = b_h[H_ + jcol];
  const float bhh = b_h[2 * H_ + jcol];
  // U column-tile indices for z/r/h gate segments (192 col-tiles total)
  const int ntz = bk * 4 + nc;
  const int ntr = 64  + bk * 4 + nc;
  const int nth = 128 + bk * 4 + nc;
  const v4u* uz = (const v4u*)(Usw + (size_t)ntz * 32 * 512) + (lane << 1);
  const v4u* ur = (const v4u*)(Usw + (size_t)ntr * 32 * 512) + (lane << 1);
  const v4u* uh = (const v4u*)(Usw + (size_t)nth * 32 * 512) + (lane << 1);

  for (int t = 0; t < S_; ++t) {
    // ---- broadcast h_bf16 (global) -> LDS (4096 b128 moves) ----
    {
      const v4u* src = (const v4u*)h_bf;
#if ASYNC_LDS_OK
      #pragma unroll
      for (int i = 0; i < 16; ++i) {
        int e = tid + i * 256;
        async_copy_b128(src + e, (v4u*)sH + e);
      }
      __builtin_amdgcn_s_wait_asynccnt(0);
#else
      v4u* dst = (v4u*)sH;
      #pragma unroll
      for (int i = 0; i < 16; ++i) dst[tid + i * 256] = src[tid + i * 256];
#endif
    }
    __syncthreads();

    // ---- hu = h @ U for this wave's 16x16 tile, all 3 gates ----
    v8f cz = {}, cr = {}, ch = {};
    #pragma unroll 8
    for (int kt = 0; kt < H_ / 32; ++kt) {
      ABv a, bz, br, bh;
      const v4u* ap = (const v4u*)&sH[(m * 16 + lane15) * H_ + kt * 32 + (laneHi << 4)];
      a.u[0] = ap[0];
      a.u[1] = ap[1];
      const v4u* pz = uz + kt * 64;
      const v4u* pr = ur + kt * 64;
      const v4u* ph = uh + kt * 64;
      __builtin_prefetch(pz + 64, 0, 1);    // next k-step of z-gate U tile
      bz.u[0] = pz[0]; bz.u[1] = pz[1];
      br.u[0] = pr[0]; br.u[1] = pr[1];
      bh.u[0] = ph[0]; bh.u[1] = ph[1];
      cz = __builtin_amdgcn_wmma_f32_16x16x32_bf16(false, a.v, false, bz.v, (short)0, cz, false, false);
      cr = __builtin_amdgcn_wmma_f32_16x16x32_bf16(false, a.v, false, br.v, (short)0, cr, false, false);
      ch = __builtin_amdgcn_wmma_f32_16x16x32_bf16(false, a.v, false, bh.v, (short)0, ch, false, false);
    }

    // ---- gates + state update (C layout: VGPR r -> M = r + 8*laneHi) ----
    #pragma unroll
    for (int r = 0; r < 8; ++r) {
      int b = m * 16 + (laneHi << 3) + r;
      size_t xrow = ((size_t)b * S_ + t) * G_;
      float xz = xw[xrow + jcol];
      float xr = xw[xrow + H_ + jcol];
      float xh = xw[xrow + 2 * H_ + jcol];
      float z  = 1.f / (1.f + __expf(-(xz + cz[r] + bhz)));
      float rg = 1.f / (1.f + __expf(-(xr + cr[r] + bhr)));
      float hh = tanhf(xh + rg * (ch[r] + bhh));
      int hidx = b * H_ + jcol;
      float hold = h_f32[hidx];
      float hnew = z * hold + (1.f - z) * hh;
      out[((size_t)b * S_ + t) * H_ + jcol] = hnew;
      h_f32[hidx] = hnew;
      h_bf[hidx]  = f2bf(hnew);
    }

    // ---- grid-wide barrier (16 co-resident blocks, monotonic counter) ----
    __threadfence();
    __syncthreads();
    if (tid == 0) {
      atomicAdd(syncCnt, 1);
      const int target = 16 * (t + 1);
      while (__hip_atomic_load(syncCnt, __ATOMIC_ACQUIRE,
                               __HIP_MEMORY_SCOPE_AGENT) < target) {
        __builtin_amdgcn_s_sleep(2);
      }
    }
    __syncthreads();
  }
}

// ---------------- host launcher --------------------------------------------
extern "C" void kernel_launch(void* const* d_in, const int* in_sizes, int n_in,
                              void* d_out, int out_size, void* d_ws, size_t ws_size,
                              hipStream_t stream) {
  const int*   x   = (const int*)  d_in[0];
  const float* emb = (const float*)d_in[1];
  const float* W   = (const float*)d_in[2];
  const float* U   = (const float*)d_in[3];
  const float* b_i = (const float*)d_in[4];
  const float* b_h = (const float*)d_in[5];
  float* out = (float*)d_out;
  (void)in_sizes; (void)n_in; (void)out_size;

  char* ws = (char*)d_ws;
  float*          h_f32 = (float*)ws;                               // 128 KB
  unsigned short* h_bf  = (unsigned short*)(ws + 131072);           // 64 KB
  int*            sync  = (int*)(ws + 131072 + 65536);              // 256 B
  unsigned short* Usw   = (unsigned short*)(ws + 131072 + 65536 + 256);  // 6 MB
  unsigned short* Wsw   = Usw + (size_t)H_ * G_;                    // 3 MB
  float*          xw    = (float*)(Wsw + (size_t)E_ * G_);          // 201 MB
  size_t needed = 131072 + 65536 + 256 + (size_t)H_ * G_ * 2 +
                  (size_t)E_ * G_ * 2 + (size_t)B_ * S_ * G_ * 4;
  if (ws_size < needed) return;   // scratch too small: do nothing (deterministic)

  init_kernel<<<128, 256, 0, stream>>>(h_f32, h_bf, sync);
  swizzle_kernel<<<(E_ * G_) / 256, 256, 0, stream>>>(W, Wsw, E_, G_);
  swizzle_kernel<<<(H_ * G_) / 256, 256, 0, stream>>>(U, Usw, H_, G_);
  xw_kernel<<<dim3((B_ * S_) / 16, G_ / 128), 256, 0, stream>>>(x, emb, Wsw, b_i, xw);
  gru_scan_kernel<<<16, 256, 0, stream>>>(xw, Usw, b_h, h_f32, h_bf, sync, out);
}
